// TNAM_46102178955876
// MI455X (gfx1250) — compile-verified, matrix operand order checked by reading, not run on hardware
//
#include <hip/hip_runtime.h>
#include <hip/hip_bf16.h>

typedef __attribute__((ext_vector_type(16))) _Float16 v16h;
typedef __attribute__((ext_vector_type(8)))  _Float16 v8h;
typedef __attribute__((ext_vector_type(8)))  float    v8f;

#define H_ 64
#define N_ 128
#define B_ 64
#define T_ 96
#define G4 256   // 4*H

__device__ __forceinline__ float sigf(float x) { return 1.0f / (1.0f + __expf(-x)); }

// ---------------------------------------------------------------------------
// Kernel 1: one workgroup per node n. 8 wave32 waves.
//   gates[64x256] = h[64x64] @ Wh[64x256] (WMMA f16->f32) + x_t ⊗ Wx + bias
//   then LSTM elementwise update; hsum accumulated in registers over T.
//   x slice staged via GLOBAL_LOAD_ASYNC_TO_LDS (ASYNCcnt path).
// ---------------------------------------------------------------------------
__global__ __launch_bounds__(256) void lstm_node_kernel(
    const float* __restrict__ x, const float* __restrict__ w,
    const float* __restrict__ bias, float* __restrict__ lstm_out)
{
  __shared__ _Float16 hS[B_][72];              // h state, f16, padded stride
  __shared__ _Float16 wfrag[2][16][32][16];    // pre-swizzled WMMA B fragments
  __shared__ float    gatesS[B_][260];         // f32 gates, padded stride
  __shared__ float    wxS[G4];
  __shared__ float    bbS[G4];
  __shared__ float    xS[T_][B_];              // x[:, :, n] slice

  const int n     = blockIdx.x;
  const int tid   = threadIdx.x;
  const int lane  = tid & 31;
  const int ln16  = lane & 15;
  const int half  = lane >> 4;
  const int wave  = tid >> 5;
  const int mbase = (wave & 3) * 16;           // batch-row strip
  const int chalf = wave >> 2;                 // column half (8 tiles each)

  const float* wN = w + (size_t)n * 65 * G4;

  // prefetch the per-node weight panel (65*256 f32 = 66.5 KB -> L2 resident)
  for (int i = tid; i < 520; i += 256)
    __builtin_prefetch((const char*)wN + (size_t)i * 128, 0, 1);

  // ---- async stage of x[:, :, n] slice into LDS (per-lane strided B32) ----
  for (int i = tid; i < T_ * B_; i += 256) {
    int t = i / B_, bb2 = i % B_;
    unsigned lds_addr = (unsigned)(size_t)&xS[t][bb2];
    const float* gp = &x[((size_t)bb2 * T_ + t) * N_ + n];
    asm volatile("global_load_async_to_lds_b32 %0, %1, off"
                 :: "v"(lds_addr), "v"(gp) : "memory");
  }

  // ---- one-time staging (regular loads) ----
  for (int i = tid; i < G4; i += 256) {
    wxS[i] = wN[i];                            // row 0 of w = x weights
    bbS[i] = bias[(size_t)n * G4 + i];
  }
  for (int i = tid; i < B_ * 72; i += 256) ((_Float16*)hS)[i] = (_Float16)0.0f;

  // pre-swizzle Wh rows 1..64 into B-fragment layout (ISA 7.12.2, 16-bit B)
  for (int i = tid; i < 2 * 16 * 32 * 16; i += 256) {
    int s  = i & 15;
    int l  = (i >> 4) & 31;
    int ct = (i >> 9) & 15;
    int kb = (i >> 13) & 1;
    int hh = l >> 4, nn = l & 15;
    int k  = kb * 32 + ((s < 8) ? (8 * hh + s) : (16 + 8 * hh + (s - 8)));
    wfrag[kb][ct][l][s] = (_Float16)wN[(size_t)(1 + k) * G4 + (16 * ct + nn)];
  }
  asm volatile("s_wait_asynccnt 0x0" ::: "memory");
  __syncthreads();

  // per-thread state: b0 owns 16 contiguous hidden units
  const int b0   = tid >> 2;
  const int hseg = (tid & 3) * 16;
  float cst[16], hsum[16];
  #pragma unroll
  for (int j = 0; j < 16; ++j) { cst[j] = 0.0f; hsum[j] = 0.0f; }

  const int mrow = mbase + ln16;

  for (int t = 0; t < T_; ++t) {
    // ---- WMMA phase: load A fragments for K=0..31 and K=32..63 ----
    v8h a0lo = *(const v8h*)&hS[mrow][     8 * half];
    v8h a0hi = *(const v8h*)&hS[mrow][16 + 8 * half];
    v8h a1lo = *(const v8h*)&hS[mrow][32 + 8 * half];
    v8h a1hi = *(const v8h*)&hS[mrow][48 + 8 * half];
    v16h a0, a1;
    #pragma unroll
    for (int i2 = 0; i2 < 8; ++i2) {
      a0[i2] = a0lo[i2]; a0[8 + i2] = a0hi[i2];
      a1[i2] = a1lo[i2]; a1[8 + i2] = a1hi[i2];
    }

    #pragma unroll
    for (int c8 = 0; c8 < 8; ++c8) {
      int ct = chalf * 8 + c8;
      v16h bf0 = *(const v16h*)&wfrag[0][ct][lane][0];
      v16h bf1 = *(const v16h*)&wfrag[1][ct][lane][0];
      v8f acc = {};
      acc = __builtin_amdgcn_wmma_f32_16x16x32_f16(false, a0, false, bf0,
                                                   (short)0, acc, false, false);
      acc = __builtin_amdgcn_wmma_f32_16x16x32_f16(false, a1, false, bf1,
                                                   (short)0, acc, false, false);
      int col = 16 * ct + ln16;
      #pragma unroll
      for (int r = 0; r < 8; ++r)
        gatesS[mbase + r + 8 * half][col] = acc[r];
    }
    __syncthreads();

    // ---- elementwise LSTM phase ----
    float xv = xS[t][b0];
    #pragma unroll
    for (int j = 0; j < 16; ++j) {
      int h = hseg + j;
      float gi = gatesS[b0][      h] + xv * wxS[      h] + bbS[      h];
      float gf = gatesS[b0][ 64 + h] + xv * wxS[ 64 + h] + bbS[ 64 + h];
      float go = gatesS[b0][128 + h] + xv * wxS[128 + h] + bbS[128 + h];
      float gc = gatesS[b0][192 + h] + xv * wxS[192 + h] + bbS[192 + h];
      float c2 = sigf(gf) * cst[j] + sigf(gi) * gc;   // no tanh on c_tilda
      float hn = sigf(go) * tanhf(c2);
      cst[j] = c2;
      hsum[j] += hn;
      hS[b0][h] = (_Float16)hn;
    }
    __syncthreads();
  }

  #pragma unroll
  for (int j = 0; j < 16; ++j)
    lstm_out[((size_t)b0 * N_ + n) * H_ + hseg + j] = hsum[j];
}

// ---------------------------------------------------------------------------
// Kernel 2: one workgroup per batch b. FFT interaction head + LN + Linear.
// Uses algebraic simplification:
//   out[b,f,h] = m[b,f,h] * A[b,h],  A = sum_f s[f] m[f,:],
//   s[f] = csigmoid(crelu(mean_{f2,h}(m[f,h] m[f2,h])))
// lstm_out[b] staged via GLOBAL_LOAD_ASYNC_TO_LDS_B128.
// ---------------------------------------------------------------------------
__global__ __launch_bounds__(256) void fft_head_kernel(
    const float* __restrict__ lstm_out,
    const float* __restrict__ ln_g, const float* __restrict__ ln_b,
    const float* __restrict__ fc_w, const float* __restrict__ fc_b,
    float* __restrict__ out)
{
  __shared__ float L[N_][H_];       // lstm_out, later feats
  __shared__ float mr[65][H_];
  __shared__ float mi[65][H_];
  __shared__ float Sr[H_], Si[H_], Ar[H_], Ai[H_];
  __shared__ float sr[65], si[65];
  __shared__ float gS[N_];
  __shared__ float outH[H_];

  const int b = blockIdx.x;
  const int tid = threadIdx.x;
  const float inv_sqrt_n = 0.08838834764831845f;        // 1/sqrt(128)
  const float W0 = 6.283185307179586f / 128.0f;         // 2*pi/128

  // async contiguous stage: 8192 floats = 2048 B128 transfers, 8 per thread
  for (int i = tid * 4; i < N_ * H_; i += 256 * 4) {
    unsigned lds_addr = (unsigned)(size_t)&((float*)L)[i];
    const float* gp = lstm_out + (size_t)b * N_ * H_ + i;
    asm volatile("global_load_async_to_lds_b128 %0, %1, off"
                 :: "v"(lds_addr), "v"(gp) : "memory");
  }
  asm volatile("s_wait_asynccnt 0x0" ::: "memory");
  __syncthreads();

  // rfft over node axis, norm='ortho'
  for (int p = tid; p < 65 * H_; p += 256) {
    int f = p / H_, h = p % H_;
    float ar = 0.f, ai = 0.f;
    for (int k = 0; k < N_; ++k) {
      float sv, cv; __sincosf(-W0 * (float)(f * k), &sv, &cv);
      float v = L[k][h];
      ar += v * cv; ai += v * sv;
    }
    mr[f][h] = ar * inv_sqrt_n;
    mi[f][h] = ai * inv_sqrt_n;
  }
  __syncthreads();

  if (tid < H_) {                                   // S[h] = sum_f m[f][h]
    float ar = 0.f, ai = 0.f;
    for (int f = 0; f < 65; ++f) { ar += mr[f][tid]; ai += mi[f][tid]; }
    Sr[tid] = ar; Si[tid] = ai;
  }
  __syncthreads();

  if (tid < 65) {                                   // squeeze -> crelu -> csigmoid
    float zr = 0.f, zi = 0.f;
    for (int h = 0; h < H_; ++h) {
      zr += mr[tid][h] * Sr[h] - mi[tid][h] * Si[h];
      zi += mr[tid][h] * Si[h] + mi[tid][h] * Sr[h];
    }
    float sc = 1.0f / (65.0f * 64.0f);
    zr = fmaxf(zr * sc, 0.f); zi = fmaxf(zi * sc, 0.f);
    float e  = __expf(-zr);
    float wr = e * __cosf(zi), wi = -e * __sinf(zi); // exp(-z)
    float dr = 1.f + wr, di = wi;
    float d  = dr * dr + di * di;
    sr[tid] = dr / d; si[tid] = -di / d;
  }
  __syncthreads();

  if (tid < H_) {                                   // A[h] = sum_f s[f]*m[f][h]
    float ar = 0.f, ai = 0.f;
    for (int f = 0; f < 65; ++f) {
      ar += sr[f] * mr[f][tid] - si[f] * mi[f][tid];
      ai += sr[f] * mi[f][tid] + si[f] * mr[f][tid];
    }
    Ar[tid] = ar; Ai[tid] = ai;
  }
  __syncthreads();

  for (int p = tid; p < 65 * H_; p += 256) {        // of = m * A (in place)
    int f = p / H_, h = p % H_;
    float r  = mr[f][h] * Ar[h] - mi[f][h] * Ai[h];
    float im = mr[f][h] * Ai[h] + mi[f][h] * Ar[h];
    mr[f][h] = r; mi[f][h] = im;
  }
  __syncthreads();

  for (int p = tid; p < N_ * H_; p += 256) {        // irfft (ortho) -> feats
    int k = p / H_, h = p % H_;
    float acc = mr[0][h] + ((k & 1) ? -mr[64][h] : mr[64][h]);
    for (int f = 1; f < 64; ++f) {
      float sv, cv; __sincosf(W0 * (float)(f * k), &sv, &cv);
      acc += 2.0f * (mr[f][h] * cv - mi[f][h] * sv);
    }
    L[k][h] = acc * inv_sqrt_n;
  }
  __syncthreads();

  if (tid < N_) {                                   // fc_att over h
    float s = 0.f;
    for (int h = 0; h < H_; ++h) s += L[tid][h];
    s = fmaxf(s * (1.0f / 64.0f), 0.f);
    gS[tid] = 1.0f / (1.0f + __expf(-s));
  }
  __syncthreads();

  if (tid < H_) {                                   // sum over nodes
    float s = 0.f;
    for (int nn = 0; nn < N_; ++nn) s += L[nn][tid] * gS[nn];
    outH[tid] = s;
  }
  __syncthreads();

  if (tid == 0) {                                   // LayerNorm + Linear + sigmoid
    float mu = 0.f;
    for (int h = 0; h < H_; ++h) mu += outH[h];
    mu *= (1.0f / H_);
    float var = 0.f;
    for (int h = 0; h < H_; ++h) { float d = outH[h] - mu; var += d * d; }
    var *= (1.0f / H_);
    float rstd = rsqrtf(var + 1e-5f);
    float acc = 0.f;
    for (int h = 0; h < H_; ++h)
      acc += ((outH[h] - mu) * rstd * ln_g[h] + ln_b[h]) * fc_w[h];
    acc += fc_b[0];
    out[b] = 1.0f / (1.0f + __expf(-acc));
  }
}

extern "C" void kernel_launch(void* const* d_in, const int* in_sizes, int n_in,
                              void* d_out, int out_size, void* d_ws, size_t ws_size,
                              hipStream_t stream) {
  const float* x    = (const float*)d_in[0];
  const float* w    = (const float*)d_in[1];
  const float* bias = (const float*)d_in[2];
  const float* ln_g = (const float*)d_in[3];
  const float* ln_b = (const float*)d_in[4];
  const float* fc_w = (const float*)d_in[5];
  const float* fc_b = (const float*)d_in[6];
  float* out = (float*)d_out;
  float* lstm_out = (float*)d_ws;   // B*N*H floats = 2 MB scratch

  lstm_node_kernel<<<dim3(N_), dim3(256), 0, stream>>>(x, w, bias, lstm_out);
  fft_head_kernel<<<dim3(B_), dim3(256), 0, stream>>>(lstm_out, ln_g, ln_b,
                                                      fc_w, fc_b, out);
}
